// SlotToImageAttention_62130996904216
// MI455X (gfx1250) — matrix operand order repaired and checked by reading, not compiled, not run on hardware
//
#include <hip/hip_runtime.h>

typedef __attribute__((ext_vector_type(16))) _Float16 v16h;
typedef __attribute__((ext_vector_type(8)))  _Float16 v8h;
typedef __attribute__((ext_vector_type(4)))  _Float16 v4h;
typedef __attribute__((ext_vector_type(8)))  float    v8f;

#define DM    256
#define EM    512
#define HEADS 8
#define HD    64
#define MQ    900
#define NKV   2048
#define BB    8

// ---------------- WMMA helpers (layouts per CDNA5 ISA 7.12.2) ----------------

// 16-bit A-matrix 16x32: lanes 0-15 hold K 0..7 (V0-3) and K 16..23 (V4-7),
// lanes 16-31 hold K 8..15 and K 24..31. LDS tile is row-major (M x K).
__device__ __forceinline__ v16h load_a_frag(const _Float16* base, int stride, int lane) {
  const int r  = lane & 15;
  const int ko = (lane >> 4) << 3;           // 0 or 8
  const _Float16* p = base + r * stride + ko;
  v16h f;
#pragma unroll
  for (int i = 0; i < 8; ++i) { f[i] = p[i]; f[i + 8] = p[i + 16]; }
  return f;
}

// 16-bit B-matrix 32x16: lane n (0-15) holds K 0..7 and K 16..23, lane n+16
// holds K 8..15 and K 24..31 (same k order as the A fragment).
// LDS tile rows are indexed by output column, 32 K-halves per row.
__device__ __forceinline__ v16h load_b_frag(const _Float16* base, int stride, int lane) {
  const int r  = lane & 15;
  const int ko = (lane >> 4) << 3;           // 0 or 8
  const _Float16* p = base + r * stride + ko;
  v16h f;
#pragma unroll
  for (int i = 0; i < 8; ++i) { f[i] = p[i]; f[i + 8] = p[i + 16]; }
  return f;
}

__device__ __forceinline__ v8f wmma_f16(v16h a, v16h b, v8f c) {
  return __builtin_amdgcn_wmma_f32_16x16x32_f16(false, a, false, b, (short)0, c,
                                                false, false);
}

__device__ __forceinline__ v4h cvt4(float4 f) {
  v4h h;
  h[0] = (_Float16)f.x; h[1] = (_Float16)f.y;
  h[2] = (_Float16)f.z; h[3] = (_Float16)f.w;
  return h;
}

// ---- CDNA5 async global->LDS copy (16B per lane, tracked on ASYNCcnt) ------
// VDST VGPR holds the workgroup-relative LDS byte offset (= low 32 bits of a
// generic pointer into a __shared__ array); hardware adds the wave's LDS base.
__device__ __forceinline__ void async_load_b128(unsigned ldsOff, const void* gptr) {
  unsigned long long ga = (unsigned long long)(uintptr_t)gptr;
  asm volatile("global_load_async_to_lds_b128 %0, %1, off"
               :: "v"(ldsOff), "v"(ga) : "memory");
}
__device__ __forceinline__ void wait_asynccnt0() {
  asm volatile("s_wait_asynccnt 0" ::: "memory");
}

// ---------------- Generic GEMM: C = alpha*(A @ W^T + bias) (+C) --------------
// A: (Mrows x K) f32 row-major.  W: (Nn x K) f32 row-major (torch linear).
// Block tile 128x128, 8 waves (4 in M x 2 in N), each wave 32x64 (2x4 WMMA).
// COLMAP remaps output columns for per-head interleave:
//   0: identity    1: col -> h*64+j (first half)    2: col -> h*64+32+j
//   3: write value to BOTH halves (used for the ksn broadcast into k)
// OUTH=1: store output as f16 (only with COLMAP=0, ACCUM=0) for attention.

#define GBM 128
#define GBN 128
#define GBK 32
#define GLDA (GBK + 16)   // 48 halves = 96B rows (16B aligned)

template <int COLMAP, int ACCUM, int OUTH>
__global__ void __launch_bounds__(256) gemm_xwt_kernel(
    const float* __restrict__ A, const float* __restrict__ W,
    const float* __restrict__ bias, void* __restrict__ Cout,
    int Mrows, int K, int ldc, float alpha)
{
  __shared__ _Float16 lA[GBM * GLDA];
  __shared__ _Float16 lB[GBN * GLDA];
  const int tid  = threadIdx.x;
  const int lane = tid & 31;
  const int wave = tid >> 5;
  const int wm = wave & 3;
  const int wn = wave >> 2;
  const int row0 = blockIdx.x * GBM;
  const int col0 = blockIdx.y * GBN;
  float* Cf = (float*)Cout;
  _Float16* Ch = (_Float16*)Cout;

  v8f acc[2][4];
#pragma unroll
  for (int i = 0; i < 2; ++i)
#pragma unroll
    for (int j = 0; j < 4; ++j)
#pragma unroll
      for (int e = 0; e < 8; ++e) acc[i][j][e] = 0.f;

  for (int k0 = 0; k0 < K; k0 += GBK) {
    // stage A tile: 1024 float4 chunks, vector load + 8B LDS store
#pragma unroll
    for (int it = 0; it < 4; ++it) {
      int chunk = tid + it * 256;
      int r = chunk >> 3;              // 8 float4 per 32-wide row
      int c4 = (chunk & 7) << 2;
      int gr = row0 + r;
      float4 f = make_float4(0.f, 0.f, 0.f, 0.f);
      if (gr < Mrows) f = *(const float4*)&A[(size_t)gr * K + k0 + c4];
      *(v4h*)&lA[r * GLDA + c4] = cvt4(f);
    }
    // stage B tile (W rows are output columns; Nn is a multiple of 128)
#pragma unroll
    for (int it = 0; it < 4; ++it) {
      int chunk = tid + it * 256;
      int r = chunk >> 3;
      int c4 = (chunk & 7) << 2;
      float4 f = *(const float4*)&W[(size_t)(col0 + r) * K + k0 + c4];
      *(v4h*)&lB[r * GLDA + c4] = cvt4(f);
    }
    __syncthreads();
    v16h af[2], bf[4];
#pragma unroll
    for (int i = 0; i < 2; ++i)
      af[i] = load_a_frag(&lA[(wm * 32 + i * 16) * GLDA], GLDA, lane);
#pragma unroll
    for (int j = 0; j < 4; ++j)
      bf[j] = load_b_frag(&lB[(wn * 64 + j * 16) * GLDA], GLDA, lane);
#pragma unroll
    for (int i = 0; i < 2; ++i)
#pragma unroll
      for (int j = 0; j < 4; ++j)
        acc[i][j] = wmma_f16(af[i], bf[j], acc[i][j]);
    __syncthreads();
  }

  // epilogue: C layout => VGPR e at lane l holds (row = e + 8*(l>=16), col = l&15)
  const int cl = lane & 15;
  const int rb = (lane >> 4) * 8;
  float bcol[4];
  int ccol[4], ccol2[4];
#pragma unroll
  for (int j = 0; j < 4; ++j) {
    int col = col0 + wn * 64 + j * 16 + cl;
    bcol[j] = bias[col];
    if (COLMAP == 0) {
      ccol[j] = col; ccol2[j] = col;
    } else {
      int hh = col >> 5, jj = col & 31;
      ccol[j]  = hh * 64 + jj + ((COLMAP == 2) ? 32 : 0);
      ccol2[j] = hh * 64 + jj + 32;
    }
  }
  const bool fullTile = (row0 + GBM) <= Mrows;
#pragma unroll
  for (int i = 0; i < 2; ++i)
#pragma unroll
    for (int e = 0; e < 8; ++e) {
      int row = row0 + wm * 32 + i * 16 + rb + e;
      if (fullTile || row < Mrows) {
        size_t rbase = (size_t)row * ldc;
#pragma unroll
        for (int j = 0; j < 4; ++j) {
          float v = (acc[i][j][e] + bcol[j]) * alpha;
          size_t off = rbase + ccol[j];
          if (OUTH) {
            Ch[off] = (_Float16)v;
          } else {
            if (ACCUM) v += Cf[off];
            Cf[off] = v;
            if (COLMAP == 3) Cf[rbase + ccol2[j]] = v;
          }
        }
      }
    }
}

// ------------- Flash attention: (B,H) x 64-row Q tiles ----------------------
// q2/k2/v2 are f16, (rows=seq*B, E) with row = seq*B + b, head slice at h*64.
// o written f32 in the same (M,B,E) layout for the final out-projection.

__global__ void __launch_bounds__(256) attn_kernel(
    const _Float16* __restrict__ q2, const _Float16* __restrict__ k2,
    const _Float16* __restrict__ v2, const unsigned char* __restrict__ amask,
    const unsigned char* __restrict__ kpm, float* __restrict__ o)
{
  __shared__ _Float16 sQ [64 * 72];
  __shared__ _Float16 sK [64 * 72];
  __shared__ _Float16 sVt[64 * 72];   // transposed: row = d, col = n-local
  __shared__ _Float16 sP [64 * 72];
  __shared__ float sS[64 * 64];
  __shared__ unsigned char sMask[64 * 64];
  __shared__ float sM[64], sL[64], sScale[64];
  __shared__ float sRed[64 * 4];

  const int tid  = threadIdx.x;
  const int lane = tid & 31;
  const int wave = tid >> 5;
  const int wm = wave & 3;        // M sub-tile (16 rows)
  const int wn = wave >> 2;       // 32-col sub-block
  const int m0 = blockIdx.x * 64;
  const int b  = blockIdx.y >> 3;
  const int h  = blockIdx.y & 7;
  const int cl = lane & 15;
  const int rb = (lane >> 4) * 8;

  // stage Q tile once (f16 copy, zero-fill padded rows)
#pragma unroll
  for (int it = 0; it < 2; ++it) {
    int chunk = tid + it * 256;
    int r = chunk >> 3;               // 8 v8h per 64-half row
    int c8 = (chunk & 7) << 3;
    int m = m0 + r;
    if (m < MQ) {
      *(v8h*)&sQ[r * 72 + c8] =
          *(const v8h*)&q2[((size_t)(m * BB + b)) * EM + h * HD + c8];
    } else {
      v8h z;
#pragma unroll
      for (int i = 0; i < 8; ++i) z[i] = (_Float16)0.f;
      *(v8h*)&sQ[r * 72 + c8] = z;
    }
  }
  if (tid < 64) { sM[tid] = -1e30f; sL[tid] = 0.f; }
  __syncthreads();

  v8f oacc[2];
#pragma unroll
  for (int j = 0; j < 2; ++j)
#pragma unroll
    for (int e = 0; e < 8; ++e) oacc[j][e] = 0.f;

  for (int n0 = 0; n0 < NKV; n0 += 64) {
    // K tile: async global->LDS (16B/lane); V tile: transposed regular stores
#pragma unroll
    for (int it = 0; it < 2; ++it) {
      int chunk = tid + it * 256;
      int r = chunk >> 3;
      int c8 = (chunk & 7) << 3;
      size_t g = ((size_t)((n0 + r) * BB + b)) * EM + h * HD + c8;
      async_load_b128((unsigned)(uintptr_t)&sK[r * 72 + c8], &k2[g]);
      v8h vv = *(const v8h*)&v2[g];
#pragma unroll
      for (int i = 0; i < 8; ++i) sVt[(c8 + i) * 72 + r] = vv[i];
    }
    // mask tile: 4 bytes of (attn_mask | key_padding_mask) per chunk
#pragma unroll
    for (int it = 0; it < 4; ++it) {
      int chunk = tid + it * 256;
      int rl = chunk >> 4;
      int c4 = (chunk & 15) << 2;
      int gm = m0 + rl;
      unsigned int am = 0u;
      if (gm < MQ) am = *(const unsigned int*)&amask[(size_t)gm * NKV + n0 + c4];
      unsigned int kp = *(const unsigned int*)&kpm[b * NKV + n0 + c4];
      *(unsigned int*)&sMask[rl * 64 + c4] = am | kp;
    }
    wait_asynccnt0();
    __syncthreads();

    // S = Q @ K^T (this 64x64 tile), masked, to LDS
    v16h aq0 = load_a_frag(&sQ[(wm * 16) * 72 + 0],  72, lane);
    v16h aq1 = load_a_frag(&sQ[(wm * 16) * 72 + 32], 72, lane);
#pragma unroll
    for (int j = 0; j < 2; ++j) {
      v16h bk0 = load_b_frag(&sK[(wn * 32 + j * 16) * 72 + 0],  72, lane);
      v16h bk1 = load_b_frag(&sK[(wn * 32 + j * 16) * 72 + 32], 72, lane);
      v8f s;
#pragma unroll
      for (int e = 0; e < 8; ++e) s[e] = 0.f;
      s = wmma_f16(aq0, bk0, s);
      s = wmma_f16(aq1, bk1, s);
#pragma unroll
      for (int e = 0; e < 8; ++e) {
        int rl = wm * 16 + rb + e;
        int nl = wn * 32 + j * 16 + cl;
        sS[rl * 64 + nl] = sMask[rl * 64 + nl] ? -1e9f : s[e];
      }
    }
    __syncthreads();

    // online softmax: 4 threads per row
    {
      int row = tid >> 2, qo = (tid & 3) * 16;
      float mx = -1e30f;
#pragma unroll
      for (int c = 0; c < 16; ++c) mx = fmaxf(mx, sS[row * 64 + qo + c]);
      sRed[row * 4 + (tid & 3)] = mx;
    }
    __syncthreads();
    if (tid < 64) {
      float mx = fmaxf(fmaxf(sRed[tid * 4 + 0], sRed[tid * 4 + 1]),
                       fmaxf(sRed[tid * 4 + 2], sRed[tid * 4 + 3]));
      float mold = sM[tid];
      float mnew = fmaxf(mold, mx);
      sScale[tid] = __expf(mold - mnew);
      sM[tid] = mnew;
    }
    __syncthreads();
    {
      int row = tid >> 2, qo = (tid & 3) * 16;
      float mnew = sM[row];
      float sum = 0.f;
#pragma unroll
      for (int c = 0; c < 16; ++c) {
        float p = __expf(sS[row * 64 + qo + c] - mnew);
        sum += p;
        sP[row * 72 + qo + c] = (_Float16)p;
      }
      sRed[row * 4 + (tid & 3)] = sum;
    }
    __syncthreads();
    if (tid < 64) {
      float s4 = sRed[tid * 4 + 0] + sRed[tid * 4 + 1] +
                 sRed[tid * 4 + 2] + sRed[tid * 4 + 3];
      sL[tid] = sL[tid] * sScale[tid] + s4;
    }
    __syncthreads();

    // rescale O accumulators (vector-load the 8 per-row factors), then P @ V
    {
      float4 sc0 = *(const float4*)&sScale[wm * 16 + rb];
      float4 sc1 = *(const float4*)&sScale[wm * 16 + rb + 4];
      float scl[8] = {sc0.x, sc0.y, sc0.z, sc0.w, sc1.x, sc1.y, sc1.z, sc1.w};
#pragma unroll
      for (int j = 0; j < 2; ++j)
#pragma unroll
        for (int e = 0; e < 8; ++e) oacc[j][e] *= scl[e];
    }
    v16h ap0 = load_a_frag(&sP[(wm * 16) * 72 + 0],  72, lane);
    v16h ap1 = load_a_frag(&sP[(wm * 16) * 72 + 32], 72, lane);
#pragma unroll
    for (int j = 0; j < 2; ++j) {
      v16h bv0 = load_b_frag(&sVt[(wn * 32 + j * 16) * 72 + 0],  72, lane);
      v16h bv1 = load_b_frag(&sVt[(wn * 32 + j * 16) * 72 + 32], 72, lane);
      oacc[j] = wmma_f16(ap0, bv0, oacc[j]);
      oacc[j] = wmma_f16(ap1, bv1, oacc[j]);
    }
    __syncthreads();
  }

  {
    float4 l0 = *(const float4*)&sL[wm * 16 + rb];
    float4 l1 = *(const float4*)&sL[wm * 16 + rb + 4];
    float lv[8] = {l0.x, l0.y, l0.z, l0.w, l1.x, l1.y, l1.z, l1.w};
#pragma unroll
    for (int j = 0; j < 2; ++j)
#pragma unroll
      for (int e = 0; e < 8; ++e) {
        int rl = wm * 16 + rb + e;
        int m = m0 + rl;
        if (m < MQ) {
          int d = wn * 32 + j * 16 + cl;
          o[((size_t)(m * BB + b)) * EM + h * HD + d] = oacc[j][e] / lv[e];
        }
      }
  }
}

// --------------------------------- driver -----------------------------------

extern "C" void kernel_launch(void* const* d_in, const int* in_sizes, int n_in,
                              void* d_out, int out_size, void* d_ws, size_t ws_size,
                              hipStream_t stream)
{
  const float* query_content = (const float*)d_in[0];
  const float* query_pos     = (const float*)d_in[1];
  const float* query_sine    = (const float*)d_in[2];
  const float* key_content   = (const float*)d_in[3];
  /* d_in[4] key_pos unused by the reference */
  const float* key_sine      = (const float*)d_in[5];
  const unsigned char* amask = (const unsigned char*)d_in[6];
  const unsigned char* kpm   = (const unsigned char*)d_in[7];
  const float* wqc = (const float*)d_in[8];  const float* bqc = (const float*)d_in[9];
  const float* wqp = (const float*)d_in[10]; const float* bqp = (const float*)d_in[11];
  const float* wqs = (const float*)d_in[12]; const float* bqs = (const float*)d_in[13];
  const float* wkc = (const float*)d_in[14]; const float* bkc = (const float*)d_in[15];
  const float* wkp = (const float*)d_in[16]; const float* bkp = (const float*)d_in[17];
  const float* wv  = (const float*)d_in[18]; const float* bv  = (const float*)d_in[19];
  const float* q_proj_w  = (const float*)d_in[20];
  const float* k_proj_w  = (const float*)d_in[21];
  const float* v_proj_w  = (const float*)d_in[22];
  const float* in_proj_b = (const float*)d_in[23];
  const float* out_w     = (const float*)d_in[24];
  const float* out_b     = (const float*)d_in[25];
  float* out = (float*)d_out;

  const int KB_ROWS = NKV * BB;   // 16384
  const int QB_ROWS = MQ * BB;    // 7200

  float* ws = (float*)d_ws;
  float* bufK  = ws;                                   // 16384*512 f32
  float* bufV  = bufK  + (size_t)KB_ROWS * EM;         // 16384*256 f32
  float* bufQ  = bufV  + (size_t)KB_ROWS * DM;         // 7200*512  f32
  _Float16* bufK2 = (_Float16*)(bufQ + (size_t)QB_ROWS * EM); // 16384*512 f16
  _Float16* bufV2 = bufK2 + (size_t)KB_ROWS * EM;             // 16384*512 f16
  _Float16* bufQ2 = bufV2 + (size_t)KB_ROWS * EM;             // 7200*512  f16
  float* bufO  = bufQ;  // q consumed once q2 is built; reuse for attention out

  dim3 blk(256);
  auto grid_of = [](int Mrows, int Nn) {
    return dim3((Mrows + GBM - 1) / GBM, Nn / GBN);
  };

  // --- projections with fused per-head interleave ---
  // k second half = ksn; first half initialized to ksn (COLMAP=3) then
  // accumulated with lin(key_content, wkc) (COLMAP=1, ACCUM=1).
  gemm_xwt_kernel<3, 0, 0><<<grid_of(KB_ROWS, DM), blk, 0, stream>>>(
      key_sine, wkp, bkp, bufK, KB_ROWS, DM, EM, 1.f);
  gemm_xwt_kernel<1, 1, 0><<<grid_of(KB_ROWS, DM), blk, 0, stream>>>(
      key_content, wkc, bkc, bufK, KB_ROWS, DM, EM, 1.f);
  gemm_xwt_kernel<0, 0, 0><<<grid_of(KB_ROWS, DM), blk, 0, stream>>>(
      key_content, wv, bv, bufV, KB_ROWS, DM, DM, 1.f);

  // q first half = lin(qc,wqc) + lin(qp,wqp); second half = lin(qs,wqs)
  gemm_xwt_kernel<1, 0, 0><<<grid_of(QB_ROWS, DM), blk, 0, stream>>>(
      query_content, wqc, bqc, bufQ, QB_ROWS, DM, EM, 1.f);
  gemm_xwt_kernel<1, 1, 0><<<grid_of(QB_ROWS, DM), blk, 0, stream>>>(
      query_pos, wqp, bqp, bufQ, QB_ROWS, DM, EM, 1.f);
  gemm_xwt_kernel<2, 0, 0><<<grid_of(QB_ROWS, DM), blk, 0, stream>>>(
      query_sine, wqs, bqs, bufQ, QB_ROWS, DM, EM, 1.f);

  // --- MHA in-projections, f16 outputs for attention ---
  // q scaled by hd^-0.5 = 0.125 (bias included, then scaled)
  gemm_xwt_kernel<0, 0, 1><<<grid_of(QB_ROWS, EM), blk, 0, stream>>>(
      bufQ, q_proj_w, in_proj_b, bufQ2, QB_ROWS, EM, EM, 0.125f);
  gemm_xwt_kernel<0, 0, 1><<<grid_of(KB_ROWS, EM), blk, 0, stream>>>(
      bufK, k_proj_w, in_proj_b + EM, bufK2, KB_ROWS, EM, EM, 1.f);
  gemm_xwt_kernel<0, 0, 1><<<grid_of(KB_ROWS, EM), blk, 0, stream>>>(
      bufV, v_proj_w, in_proj_b + 2 * EM, bufV2, KB_ROWS, DM, EM, 1.f);

  // --- fused masked-softmax attention -> o in (M,B,E) layout ---
  dim3 agrid((MQ + 63) / 64, BB * HEADS);
  attn_kernel<<<agrid, blk, 0, stream>>>(bufQ2, bufK2, bufV2, amask, kpm, bufO);

  // --- out-projection ---
  gemm_xwt_kernel<0, 0, 0><<<grid_of(QB_ROWS, EM), blk, 0, stream>>>(
      bufO, out_w, out_b, out, QB_ROWS, EM, EM, 1.f);

  (void)in_sizes; (void)n_in; (void)out_size; (void)ws_size;
}
